// RGAT_74431783240014
// MI455X (gfx1250) — matrix operand order repaired and checked by reading, not compiled, not run on hardware
//
#include <hip/hip_runtime.h>
#include <hip/hip_bf16.h>

typedef __attribute__((ext_vector_type(16))) __bf16        v16bf;
typedef __attribute__((ext_vector_type(8)))  float         v8f;
typedef __attribute__((ext_vector_type(4)))  unsigned int  u32x4;

#define DD        128
#define NCOL_NODE 768   // [S | Dd | Sq | Dq | LOOP | EVO] each 128 cols
#define NCOL_REL  256   // [R | Rq]

// ---------------------------------------------------------------- helpers

__device__ __forceinline__ unsigned short f32_to_bf16(float f) {
  unsigned int u = __float_as_uint(f);
  unsigned int r = 0x7FFFu + ((u >> 16) & 1u);   // round-to-nearest-even
  return (unsigned short)((u + r) >> 16);
}

// A fragment (16x32 bf16): lane(0-15)=row M, elems 0..7 = K+0..7, elems 8..15 = K+16..23
__device__ __forceinline__ v16bf load_frag_a(const unsigned short* p) {
  union { v16bf v; u32x4 u[2]; } t;
  t.u[0] = *(const u32x4*)p;
  t.u[1] = *(const u32x4*)(p + 16);
  return t.v;
}
// B fragment (32x16 bf16, column-major source): lane(0-15)=col N, elems 0..7 = K+0..7, elems 8..15 = K+16..23
__device__ __forceinline__ v16bf load_frag_b(const unsigned short* p) {
  union { v16bf v; u32x4 u[2]; } t;
  t.u[0] = *(const u32x4*)p;
  t.u[1] = *(const u32x4*)(p + 16);
  return t.v;
}

__device__ __forceinline__ void atomicMaxF(float* addr, float v) {
  // Valid for buffers initialized to -inf; handles mixed signs.
  if (v >= 0.0f) atomicMax((int*)addr, __float_as_int(v));
  else           atomicMin((unsigned int*)addr, __float_as_uint(v));
}

__device__ __forceinline__ float leaky(float x) { return x > 0.0f ? x : 0.01f * x; }

// ---------------------------------------------------------------- K0: build transposed bf16 weight panels
// BtN[c][k] (c<768), BtR[c][k] (c<256); composed products W*@w_quad computed here.
__global__ void __launch_bounds__(256) prep_weights(
    const float* __restrict__ wt,   // (384,128) row-major: [W_s; W_r; W_d]
    const float* __restrict__ wq,   // (128,128)
    const float* __restrict__ loopw,
    const float* __restrict__ evo,
    unsigned short* __restrict__ BtN,
    unsigned short* __restrict__ BtR) {
  int idx = blockIdx.x * 256 + threadIdx.x;
  const int NODE_ELEMS = NCOL_NODE * DD;
  const int REL_ELEMS  = NCOL_REL * DD;
  if (idx < NODE_ELEMS) {
    int c = idx >> 7, k = idx & 127;
    float val;
    if (c < 128)      val = wt[k * DD + c];                    // W_s
    else if (c < 256) val = wt[(256 + k) * DD + (c - 128)];    // W_d
    else if (c < 384) {                                        // W_s @ w_quad
      int cc = c - 256; float s = 0.f;
      for (int j = 0; j < DD; ++j) s += wt[k * DD + j] * wq[j * DD + cc];
      val = s;
    } else if (c < 512) {                                      // W_d @ w_quad
      int cc = c - 384; float s = 0.f;
      for (int j = 0; j < DD; ++j) s += wt[(256 + k) * DD + j] * wq[j * DD + cc];
      val = s;
    } else if (c < 640) val = loopw[k * DD + (c - 512)];
    else                val = evo[k * DD + (c - 640)];
    BtN[(size_t)c * DD + k] = f32_to_bf16(val);
  } else if (idx < NODE_ELEMS + REL_ELEMS) {
    int l = idx - NODE_ELEMS;
    int c = l >> 7, k = l & 127;
    float val;
    if (c < 128) val = wt[(128 + k) * DD + c];                 // W_r
    else {                                                     // W_r @ w_quad
      int cc = c - 128; float s = 0.f;
      for (int j = 0; j < DD; ++j) s += wt[(128 + k) * DD + j] * wq[j * DD + cc];
      val = s;
    }
    BtR[(size_t)c * DD + k] = f32_to_bf16(val);
  }
}

// ---------------------------------------------------------------- K1: fp32 -> bf16 (with zero padding rows)
__global__ void __launch_bounds__(256) convert_bf16(
    const float* __restrict__ node, const float* __restrict__ rel,
    unsigned short* __restrict__ nodeB, unsigned short* __restrict__ relB,
    int nValid, int nTot, int rValid, int rTot) {
  int i = blockIdx.x * 256 + threadIdx.x;
  if (i < nTot) nodeB[i] = (i < nValid) ? f32_to_bf16(node[i]) : (unsigned short)0;
  int j = i - nTot;
  if (j >= 0 && j < rTot) relB[j] = (j < rValid) ? f32_to_bf16(rel[j]) : (unsigned short)0;
}

// ---------------------------------------------------------------- K2: C[M, ncols] = A[M,128](bf16) @ Bt (bf16, col-major as Bt[c][k])
// One wave -> two 16x16 output tiles (shared B fragment). M must be multiple of 32.
__global__ void __launch_bounds__(256) wmma_gemm_k128(
    const unsigned short* __restrict__ A,
    const unsigned short* __restrict__ Bt,
    float* __restrict__ C, int ncols) {
  const int lane  = threadIdx.x & 31;
  const int wave  = threadIdx.x >> 5;
  const int nBase = (blockIdx.y * 8 + wave) * 16;
  if (nBase >= ncols) return;                 // uniform per wave
  const int mBase = blockIdx.x * 32;
  const int l15 = lane & 15;
  const int hi  = lane >> 4;
  const unsigned short* pa0 = A  + (size_t)(mBase + l15) * DD + (hi << 3);
  const unsigned short* pa1 = pa0 + (size_t)16 * DD;
  const unsigned short* pb  = Bt + (size_t)(nBase + l15) * DD + (hi << 3);
  v8f acc0 = {}; v8f acc1 = {};
#pragma unroll
  for (int k0 = 0; k0 < DD; k0 += 32) {
    v16bf b  = load_frag_b(pb  + k0);
    v16bf a0 = load_frag_a(pa0 + k0);
    v16bf a1 = load_frag_a(pa1 + k0);
    acc0 = __builtin_amdgcn_wmma_f32_16x16x32_bf16(false, a0, false, b, (short)0, acc0, false, false);
    acc1 = __builtin_amdgcn_wmma_f32_16x16x32_bf16(false, a1, false, b, (short)0, acc1, false, false);
  }
  // C/D layout: VGPR r -> M = r + 8*hi, N = l15
  float* pc0 = C + (size_t)(mBase + hi * 8) * ncols + nBase + l15;
  float* pc1 = pc0 + (size_t)16 * ncols;
#pragma unroll
  for (int r = 0; r < 8; ++r) {
    pc0[(size_t)r * ncols] = acc0[r];
    pc1[(size_t)r * ncols] = acc1[r];
  }
}

// ---------------------------------------------------------------- K3: init accumulators
__global__ void __launch_bounds__(256) init_accum(
    float* __restrict__ amax, float* __restrict__ denom, float* __restrict__ numer,
    unsigned int* __restrict__ indeg, int n128, int n) {
  int i = blockIdx.x * 256 + threadIdx.x;
  if (i < n128) { amax[i] = -__builtin_inff(); denom[i] = 0.f; numer[i] = 0.f; }
  if (i < n) indeg[i] = 0u;
}

// ---------------------------------------------------------------- K4: segment max of leaky(tripletq), one wave per edge
__global__ void __launch_bounds__(256) edge_max_kernel(
    const float* __restrict__ Cn, const float* __restrict__ Cr,
    const int* __restrict__ src, const int* __restrict__ dst, const int* __restrict__ et,
    float* __restrict__ amax, int E) {
  int e = blockIdx.x * 8 + (threadIdx.x >> 5);
  if (e >= E) return;
  int lane = threadIdx.x & 31;
  int s = src[e], d = dst[e], r = et[e];
  const float4 sq = *(const float4*)(Cn + (size_t)s * NCOL_NODE + 256 + lane * 4);
  const float4 rq = *(const float4*)(Cr + (size_t)r * NCOL_REL  + 128 + lane * 4);
  const float4 dq = *(const float4*)(Cn + (size_t)d * NCOL_NODE + 384 + lane * 4);
  float* am = amax + (size_t)d * DD + lane * 4;
  float v0 = leaky(sq.x + rq.x + dq.x);
  float v1 = leaky(sq.y + rq.y + dq.y);
  float v2 = leaky(sq.z + rq.z + dq.z);
  float v3 = leaky(sq.w + rq.w + dq.w);
  atomicMaxF(am + 0, v0); atomicMaxF(am + 1, v1);
  atomicMaxF(am + 2, v2); atomicMaxF(am + 3, v3);
}

// ---------------------------------------------------------------- K5: accumulate softmax denom and numerator
__global__ void __launch_bounds__(256) edge_accum_kernel(
    const float* __restrict__ Cn, const float* __restrict__ Cr,
    const int* __restrict__ src, const int* __restrict__ dst, const int* __restrict__ et,
    const float* __restrict__ amax,
    float* __restrict__ denom, float* __restrict__ numer,
    unsigned int* __restrict__ indeg, int E) {
  int e = blockIdx.x * 8 + (threadIdx.x >> 5);
  if (e >= E) return;
  int lane = threadIdx.x & 31;
  int s = src[e], d = dst[e], r = et[e];
  const float4 st = *(const float4*)(Cn + (size_t)s * NCOL_NODE +   0 + lane * 4);
  const float4 rt = *(const float4*)(Cr + (size_t)r * NCOL_REL  +   0 + lane * 4);
  const float4 dt = *(const float4*)(Cn + (size_t)d * NCOL_NODE + 128 + lane * 4);
  const float4 sq = *(const float4*)(Cn + (size_t)s * NCOL_NODE + 256 + lane * 4);
  const float4 rq = *(const float4*)(Cr + (size_t)r * NCOL_REL  + 128 + lane * 4);
  const float4 dq = *(const float4*)(Cn + (size_t)d * NCOL_NODE + 384 + lane * 4);
  const float4 mx = *(const float4*)(amax + (size_t)d * DD + lane * 4);
  size_t off = (size_t)d * DD + lane * 4;
  float t0 = st.x + rt.x + dt.x, t1 = st.y + rt.y + dt.y;
  float t2 = st.z + rt.z + dt.z, t3 = st.w + rt.w + dt.w;
  float e0 = __expf(leaky(sq.x + rq.x + dq.x) - mx.x);
  float e1 = __expf(leaky(sq.y + rq.y + dq.y) - mx.y);
  float e2 = __expf(leaky(sq.z + rq.z + dq.z) - mx.z);
  float e3 = __expf(leaky(sq.w + rq.w + dq.w) - mx.w);
  atomicAdd(denom + off + 0, e0); atomicAdd(numer + off + 0, e0 * t0);
  atomicAdd(denom + off + 1, e1); atomicAdd(numer + off + 1, e1 * t1);
  atomicAdd(denom + off + 2, e2); atomicAdd(numer + off + 2, e2 * t2);
  atomicAdd(denom + off + 3, e3); atomicAdd(numer + off + 3, e3 * t3);
  if (lane == 0) atomicAdd(indeg + d, 1u);
}

// ---------------------------------------------------------------- K6: h + loop, then LayerNorm; one wave per node
__global__ void __launch_bounds__(256) finalize_kernel(
    const float* __restrict__ Cn, const float* __restrict__ numer,
    const float* __restrict__ denom, const unsigned int* __restrict__ indeg,
    const float* __restrict__ norm, float* __restrict__ out, int N) {
  int node = blockIdx.x * 8 + (threadIdx.x >> 5);
  if (node >= N) return;
  int lane = threadIdx.x & 31;
  unsigned deg = indeg[node];            // wave-uniform
  float o0, o1, o2, o3;
  if (deg) {
    const float4 nu = *(const float4*)(numer + (size_t)node * DD + lane * 4);
    const float4 de = *(const float4*)(denom + (size_t)node * DD + lane * 4);
    const float4 lp = *(const float4*)(Cn + (size_t)node * NCOL_NODE + 512 + lane * 4);
    float nrm = norm[node];
    o0 = nu.x / de.x * nrm + lp.x;
    o1 = nu.y / de.y * nrm + lp.y;
    o2 = nu.z / de.z * nrm + lp.z;
    o3 = nu.w / de.w * nrm + lp.w;
  } else {
    const float4 lp = *(const float4*)(Cn + (size_t)node * NCOL_NODE + 640 + lane * 4);
    o0 = lp.x; o1 = lp.y; o2 = lp.z; o3 = lp.w;
  }
  float s  = o0 + o1 + o2 + o3;
  float ss = o0 * o0 + o1 * o1 + o2 * o2 + o3 * o3;
#pragma unroll
  for (int m = 16; m > 0; m >>= 1) {
    s  += __shfl_xor(s,  m, 32);
    ss += __shfl_xor(ss, m, 32);
  }
  float mu   = s * (1.0f / 128.0f);
  float var  = ss * (1.0f / 128.0f) - mu * mu;
  float rstd = rsqrtf(var + 1e-5f);
  float4 res;
  res.x = (o0 - mu) * rstd; res.y = (o1 - mu) * rstd;
  res.z = (o2 - mu) * rstd; res.w = (o3 - mu) * rstd;
  *(float4*)(out + (size_t)node * DD + lane * 4) = res;
}

// ---------------------------------------------------------------- host
extern "C" void kernel_launch(void* const* d_in, const int* in_sizes, int n_in,
                              void* d_out, int out_size, void* d_ws, size_t ws_size,
                              hipStream_t stream) {
  const float* node  = (const float*)d_in[0];
  const float* rel   = (const float*)d_in[1];
  const float* norm  = (const float*)d_in[2];
  const float* wt    = (const float*)d_in[3];
  const float* wq    = (const float*)d_in[4];
  const float* loopw = (const float*)d_in[5];
  const float* evo   = (const float*)d_in[6];
  const int*   src   = (const int*)d_in[7];
  const int*   dst   = (const int*)d_in[8];
  const int*   et    = (const int*)d_in[9];

  const int N  = in_sizes[0] / DD;        // 50000
  const int R  = in_sizes[1] / DD;        // 200
  const int E  = in_sizes[7];             // 640000
  const int NP = (N + 31) & ~31;          // pad M for 32-row WMMA macro-tiles
  const int RP = (R + 31) & ~31;

  char* ws = (char*)d_ws;
  size_t off = 0;
  auto alloc = [&](size_t bytes) -> char* {
    char* p = ws + off;
    off += (bytes + 255) & ~(size_t)255;
    return p;
  };
  unsigned short* nodeB = (unsigned short*)alloc((size_t)NP * DD * 2);
  unsigned short* relB  = (unsigned short*)alloc((size_t)RP * DD * 2);
  unsigned short* BtN   = (unsigned short*)alloc((size_t)NCOL_NODE * DD * 2);
  unsigned short* BtR   = (unsigned short*)alloc((size_t)NCOL_REL  * DD * 2);
  float* Cn    = (float*)alloc((size_t)NP * NCOL_NODE * 4);
  float* Cr    = (float*)alloc((size_t)RP * NCOL_REL  * 4);
  float* amax  = (float*)alloc((size_t)N * DD * 4);
  float* denom = (float*)alloc((size_t)N * DD * 4);
  float* numer = (float*)alloc((size_t)N * DD * 4);
  unsigned int* indeg = (unsigned int*)alloc((size_t)N * 4);
  (void)ws_size; (void)n_in; (void)out_size;

  // K0: weight panels (incl. composed W*@w_quad), transposed bf16
  {
    int total = NCOL_NODE * DD + NCOL_REL * DD;
    prep_weights<<<(total + 255) / 256, 256, 0, stream>>>(wt, wq, loopw, evo, BtN, BtR);
  }
  // K1: activations -> bf16 with zero padding
  {
    int nTot = NP * DD, rTot = RP * DD;
    int total = nTot + rTot;
    convert_bf16<<<(total + 255) / 256, 256, 0, stream>>>(node, rel, nodeB, relB,
                                                          N * DD, nTot, R * DD, rTot);
  }
  // K2: node-level GEMM (NP x 768) and rel-level GEMM (RP x 256)
  {
    dim3 g1(NP / 32, (NCOL_NODE + 127) / 128);
    wmma_gemm_k128<<<g1, 256, 0, stream>>>(nodeB, BtN, Cn, NCOL_NODE);
    dim3 g2(RP / 32, (NCOL_REL + 127) / 128);
    wmma_gemm_k128<<<g2, 256, 0, stream>>>(relB, BtR, Cr, NCOL_REL);
  }
  // K3: init segment accumulators
  {
    int n128 = N * DD;
    init_accum<<<(n128 + 255) / 256, 256, 0, stream>>>(amax, denom, numer, indeg, n128, N);
  }
  // K4/K5: edge passes (one wave per edge)
  {
    int blocks = (E + 7) / 8;
    edge_max_kernel<<<blocks, 256, 0, stream>>>(Cn, Cr, src, dst, et, amax, E);
    edge_accum_kernel<<<blocks, 256, 0, stream>>>(Cn, Cr, src, dst, et, amax,
                                                  denom, numer, indeg, E);
  }
  // K6: finalize + LayerNorm
  {
    int blocks = (N + 7) / 8;
    finalize_kernel<<<blocks, 256, 0, stream>>>(Cn, numer, denom, indeg, norm,
                                                (float*)d_out, N);
  }
}
